// Decoder_31636729102577
// MI455X (gfx1250) — matrix operand order repaired and checked by reading, not compiled, not run on hardware
//
#include <hip/hip_runtime.h>
#include <hip/hip_fp16.h>
#include <stdint.h>

// ---------------------------------------------------------------------------
// LSTM decoder w/ attention for MI455X (gfx1250, wave32, WMMA).
//   H=E=512, V=32000, B=32, T=64, S=64
// Strategy (unchanged from round 1, plus TDM staging):
//   prep_kernel      : f32->f16 weight conversion, W2=[W_ih|W_hh], src transpose,
//                      embedding gather, state init, barrier reset.
//   gemm_f16_tn      : 128x128-tile WMMA GEMM, C=A(MxK)*B(NxK)^T.
//                      A tile: per-lane global_load_async_to_lds_b128 (ASYNCcnt)
//                      B tile: one tensor_load_to_lds TDM descriptor per stage
//                              issued by wave 0 (TENSORcnt), double buffered.
//   recurrent_kernel : persistent 8-WG kernel, global atomic barrier; per step:
//                      gates WMMA GEMM -> LSTM elementwise -> attention (VALU)
//                      -> att_vec WMMA GEMM -> tanh; stores att_t for readout.
//   readout          : (T*B=2048, 512) x (512, 32000) WMMA GEMM -> d_out.
// ---------------------------------------------------------------------------

typedef __attribute__((ext_vector_type(16))) _Float16 v16h;
typedef __attribute__((ext_vector_type(8)))  float    v8f;
typedef __attribute__((ext_vector_type(4)))  unsigned int v4u;
typedef __attribute__((ext_vector_type(8)))  unsigned int v8u;

constexpr int kH   = 512;
constexpr int kE   = 512;
constexpr int kV   = 32000;
constexpr int kB   = 32;
constexpr int kT   = 64;
constexpr int kS   = 64;
constexpr int kKih = 1536;   // [emb(512) | att(512) | h(512)]
constexpr int kG4  = 2048;   // 4H gate columns
constexpr int kNWG = 8;      // persistent workgroups for recurrence

struct Prms {
  const float* src;      // (S,B,2H)
  const float* h0;       // (1,B,H)
  const float* c0;       // (1,B,H)
  const float* embTab;   // (V,E)
  const float* Wih;      // (4H, E+H)
  const float* Whh;      // (4H, H)
  const float* bih;      // (4H)
  const float* bhh;      // (4H)
  const float* attSrcW;  // (H, 2H)
  const float* attVecW;  // (H, 3H)
  const float* rdW;      // (V, H)
  const int*   tgt;      // (T,B)
  _Float16* rdW16;       // (V,H)
  _Float16* W2;          // (4H, 1536) = [Wih | Whh]
  _Float16* attW16;      // (H, 1536)
  _Float16* attSrcW16;   // (H, 1024)
  _Float16* srcT16;      // (B*S, 1024)
  float*    srcT32;      // (B*S, 1024)
  float*    srcAtt;      // (B*S, H)
  _Float16* embF;        // (T*B, E)
  _Float16* attAll;      // (T*B, H)
  _Float16* Ag;          // (B, 1536)  gates A = [emb_t | att | h]
  _Float16* Av;          // (B, 1536)  att_vec A = [h | ctx]
  float*    hF;          // (B,H)
  float*    cF;          // (B,H)
  float*    bsum;        // (4H)
  unsigned* cnt;         // global barrier counter
  float*    out;         // (T*B, V)
};

// ---------------------------- helpers --------------------------------------

__device__ inline v8f wmma16(v16h a, v16h b, v8f c) {
  return __builtin_amdgcn_wmma_f32_16x16x32_f16(false, a, false, b,
                                                (short)0, c, false, false);
}

union FragU { v16h v; int4 q[2]; };

// A-fragment (16x32 f16). Lane L: M=L%16, half h=L/16; bytes h*16 / h*16+32.
__device__ inline v16h fragA_lds(const _Float16* s, int lane, int m0) {
  int h = (lane >> 4) & 1, m = m0 + (lane & 15);
  const char* p = (const char*)(s + m * 32) + h * 16;
  FragU u; u.q[0] = *(const int4*)p; u.q[1] = *(const int4*)(p + 32);
  return u.v;
}
__device__ inline v16h fragA_g(const _Float16* A, int lane, int m0, int k0, int ldk) {
  int h = (lane >> 4) & 1, m = m0 + (lane & 15);
  const _Float16* p = A + (size_t)m * ldk + k0 + h * 8;
  FragU u; u.q[0] = *(const int4*)p; u.q[1] = *(const int4*)(p + 16);
  return u.v;
}
// B-fragment (32x16 f16). Lane L: N=L%16, kgroup=L/16; K = kgroup*8 + e (e<8), 16 + kgroup*8 + (e-8).
__device__ inline v16h fragB_lds(const _Float16* s, int lane, int n0) {
  int kg = (lane >> 4) & 1, n = n0 + (lane & 15);
  const char* p = (const char*)(s + n * 32) + kg * 16;
  FragU u; u.q[0] = *(const int4*)p; u.q[1] = *(const int4*)(p + 32);
  return u.v;
}
__device__ inline v16h fragB_g(const _Float16* B, int lane, int n0, int k0, int ldk) {
  int kg = (lane >> 4) & 1, n = n0 + (lane & 15);
  const _Float16* p = B + (size_t)n * ldk + k0 + kg * 8;
  FragU u; u.q[0] = *(const int4*)p; u.q[1] = *(const int4*)(p + 16);
  return u.v;
}

// async global -> LDS, 16B/lane (ASYNCcnt). Low 32 bits of a generic shared
// pointer are the wave-relative LDS byte address (LDS aperture, ISA 10.2).
__device__ inline void async_ld_b128(uint32_t lds_byte, const void* g) {
  asm volatile("global_load_async_to_lds_b128 %0, %1, off"
               :: "v"(lds_byte), "v"(g) : "memory");
}
__device__ inline void wait_async_le2() { asm volatile("s_wait_asynccnt 2" ::: "memory"); }
__device__ inline void wait_async_0()   { asm volatile("s_wait_asynccnt 0" ::: "memory"); }

// TDM: load a 2D tile (tile_dim0=32 f16 elems x tile_dim1=128 rows, row stride
// = ldk elems) from global into contiguous LDS. Descriptor per ISA 08 §8.3/8.4:
//   group0: count=1, lds_addr, global_addr[56:0], type=2
//   group1: data_size=1 (2B), tensor dims huge (tiles always in-bounds),
//           tile_dim0=32, tile_dim1=128, tensor_dim0_stride=ldk.
__device__ inline void tdm_load_tile(uint32_t lds_byte, const void* g, uint32_t ldk) {
  uint64_t ga = (uint64_t)(uintptr_t)g;
  v4u g0; v8u g1;
  g0[0] = 1u;                                            // count=1, user mode
  g0[1] = lds_byte;                                      // lds_addr
  g0[2] = (uint32_t)ga;                                  // global_addr[31:0]
  g0[3] = (uint32_t)((ga >> 32) & 0x01FFFFFFu) | (2u << 30); // addr[56:32]|type=2
  g1[0] = 0x00010000u;                                   // data_size=2B
  g1[1] = 0xFFFF0000u;                                   // tensor_dim0 lo16 (0x7fffffff)
  g1[2] = 0xFFFF7FFFu;                                   // td0 hi | td1 lo
  g1[3] = 0x00207FFFu;                                   // td1 hi | tile_dim0=32
  g1[4] = 128u;                                          // tile_dim1=128, tile_dim2=0
  g1[5] = ldk;                                           // tensor_dim0_stride lo32
  g1[6] = 0u;                                            // stride hi | dim1_stride lo
  g1[7] = 0u;
  asm volatile("tensor_load_to_lds %0, %1" :: "s"(g0), "s"(g1) : "memory");
}

__device__ inline float sigf(float x) { return 1.0f / (1.0f + __expf(-x)); }

__device__ inline void grid_barrier(unsigned* cnt, unsigned target) {
  __threadfence();
  __syncthreads();
  if (threadIdx.x == 0) {
    atomicAdd(cnt, 1u);
    while (atomicAdd(cnt, 0u) < target) __builtin_amdgcn_s_sleep(1);
  }
  __syncthreads();
  __threadfence();
}

// ------------------------------ prep ---------------------------------------

__global__ __launch_bounds__(256) void prep_kernel(Prms p) {
  const int stride = gridDim.x * blockDim.x;
  const int t0 = blockIdx.x * blockDim.x + threadIdx.x;

  for (int i = t0; i < kV * kH; i += stride) p.rdW16[i] = (_Float16)p.rdW[i];
  for (int i = t0; i < kG4 * kKih; i += stride) {
    int n = i / kKih, k = i - n * kKih;
    float v = (k < 1024) ? p.Wih[n * 1024 + k] : p.Whh[n * kH + (k - 1024)];
    p.W2[i] = (_Float16)v;
  }
  for (int i = t0; i < kH * 1536; i += stride) p.attW16[i] = (_Float16)p.attVecW[i];
  for (int i = t0; i < kH * 1024; i += stride) p.attSrcW16[i] = (_Float16)p.attSrcW[i];
  for (int i = t0; i < kB * kS * 1024; i += stride) {   // (S,B,2H)->(B,S,2H)
    int k = i & 1023; int bs = i >> 10; int b = bs >> 6; int s = bs & 63;
    float v = p.src[(size_t)(s * kB + b) * 1024 + k];
    p.srcT32[i] = v; p.srcT16[i] = (_Float16)v;
  }
  for (int i = t0; i < kT * kB * kE; i += stride) {     // embedding gather
    int e = i & 511; int tb = i >> 9; int tok = p.tgt[tb];
    p.embF[i] = (_Float16)p.embTab[(size_t)tok * kE + e];
  }
  for (int i = t0; i < kG4; i += stride) p.bsum[i] = p.bih[i] + p.bhh[i];
  for (int i = t0; i < kB * kH; i += stride) {          // state init + Ag
    int b = i >> 9, j = i & 511;
    float hv = p.h0[i], cv = p.c0[i];
    p.hF[i] = hv; p.cF[i] = cv;
    p.Ag[b * kKih + j]        = (_Float16)p.embTab[(size_t)p.tgt[b] * kE + j];
    p.Ag[b * kKih + 512 + j]  = (_Float16)0.0f;
    p.Ag[b * kKih + 1024 + j] = (_Float16)hv;
  }
  if (t0 == 0) *p.cnt = 0u;
}

// --------------------- generic 128x128 WMMA GEMM ----------------------------
// C(MxN,f32) = A(MxK,f16,row) * B(NxK,f16,row)^T.  M,N mult of 128; K of 32.
// grid=(N/128, M/128), 256 threads (8 waves, 4x2), wave tile 32x64.

__global__ __launch_bounds__(256) void gemm_f16_tn(const _Float16* __restrict__ A,
                                                   const _Float16* __restrict__ B,
                                                   float* __restrict__ C,
                                                   int M, int N, int K) {
  __shared__ __align__(16) _Float16 sA[2][128 * 32];
  __shared__ __align__(16) _Float16 sB[2][128 * 32];
  const int tid = threadIdx.x, lane = tid & 31, wave = tid >> 5;
  const int mt2 = wave & 3, nt2 = wave >> 2;
  const int bm = blockIdx.y * 128, bn = blockIdx.x * 128;
  const _Float16* Ab = A + (size_t)bm * K;
  const _Float16* Bb = B + (size_t)bn * K;

  auto issueA = [&](int buf, int k0) {       // 8KB A: 2 async b128 per thread
#pragma unroll
    for (int i = 0; i < 2; i++) {
      int off = i * 4096 + tid * 16;
      int row = off >> 6, colb = off & 63;
      async_ld_b128((uint32_t)(uintptr_t)&sA[buf][0] + off,
                    (const char*)(Ab + (size_t)row * K + k0) + colb);
    }
  };
  auto issueB = [&](int buf, int k0) {       // 8KB B: one TDM descriptor (wave 0)
    tdm_load_tile((uint32_t)(uintptr_t)&sB[buf][0], Bb + k0, (uint32_t)K);
  };

  v8f acc[2][4] = {};
  const int nk = K / 32;
  issueA(0, 0);
  if (wave == 0) issueB(0, 0);
  for (int ks = 0; ks < nk; ks++) {
    if (ks + 1 < nk) {
      issueA((ks + 1) & 1, (ks + 1) * 32);
      if (wave == 0) issueB((ks + 1) & 1, (ks + 1) * 32);
      wait_async_le2();
      if (wave == 0) __builtin_amdgcn_s_wait_tensorcnt(1);
    } else {
      wait_async_0();
      if (wave == 0) __builtin_amdgcn_s_wait_tensorcnt(0);
    }
    __syncthreads();
    const _Float16* pa = sA[ks & 1];
    const _Float16* pb = sB[ks & 1];
    v16h a0 = fragA_lds(pa, lane, mt2 * 32);
    v16h a1 = fragA_lds(pa, lane, mt2 * 32 + 16);
    v16h bf[4];
#pragma unroll
    for (int j = 0; j < 4; j++) bf[j] = fragB_lds(pb, lane, nt2 * 64 + j * 16);
#pragma unroll
    for (int j = 0; j < 4; j++) {
      acc[0][j] = wmma16(a0, bf[j], acc[0][j]);
      acc[1][j] = wmma16(a1, bf[j], acc[1][j]);
    }
    __syncthreads();
  }
#pragma unroll
  for (int mi = 0; mi < 2; mi++)
#pragma unroll
    for (int j = 0; j < 4; j++)
#pragma unroll
      for (int r = 0; r < 8; r++) {
        int row = bm + mt2 * 32 + mi * 16 + ((lane >> 4) & 1) * 8 + r;
        int col = bn + nt2 * 64 + j * 16 + (lane & 15);
        C[(size_t)row * N + col] = acc[mi][j][r];
      }
}

// ------------------------- persistent recurrence ----------------------------

__global__ __launch_bounds__(256) void recurrent_kernel(Prms p) {
  const int wg = blockIdx.x, tid = threadIdx.x, lane = tid & 31, wave = tid >> 5;
  __shared__ float gbuf[kB * 256];   // this WG's gate slice (32 x 256)
  __shared__ float sl[kB * kS];      // attention logits / weights (WG0)
  unsigned epoch = 0;
  const int j0 = wg * 64;            // hidden-slice owned by this WG

  for (int t = 0; t < kT; t++) {
    // ---- Phase A: gates slice = Ag(32x1536) * W2[n-slice]^T ----
    {
      const int mt = wave & 1, ng = wave >> 1;
      v8f acc[4] = {};
      for (int k0 = 0; k0 < kKih; k0 += 32) {
        v16h a = fragA_g(p.Ag, lane, mt * 16, k0, kKih);
        v16h bf[4];
#pragma unroll
        for (int i = 0; i < 4; i++) {
          int nt = ng * 4 + i, gate = nt >> 2, jt = nt & 3;
          bf[i] = fragB_g(p.W2, lane, gate * 512 + j0 + jt * 16, k0, kKih);
        }
#pragma unroll
        for (int i = 0; i < 4; i++) acc[i] = wmma16(a, bf[i], acc[i]);
      }
#pragma unroll
      for (int i = 0; i < 4; i++) {
        int nt = ng * 4 + i, gate = nt >> 2, jt = nt & 3;
        int l0 = gate * 64 + jt * 16 + (lane & 15);
        int rowb = mt * 16 + ((lane >> 4) & 1) * 8;
#pragma unroll
        for (int r = 0; r < 8; r++) gbuf[(rowb + r) * 256 + l0] = acc[i][r];
      }
    }
    __syncthreads();
    // ---- LSTM elementwise on owned slice ----
    for (int e = tid; e < kB * 64; e += 256) {
      int b = e >> 6, jj = e & 63, j = j0 + jj;
      float gi = gbuf[b * 256 + jj]       + p.bsum[j];
      float gf = gbuf[b * 256 + 64 + jj]  + p.bsum[512 + j];
      float gg = gbuf[b * 256 + 128 + jj] + p.bsum[1024 + j];
      float go = gbuf[b * 256 + 192 + jj] + p.bsum[1536 + j];
      float cO = p.cF[b * kH + j];
      float cn = sigf(gf) * cO + sigf(gi) * tanhf(gg);
      float hn = sigf(go) * tanhf(cn);
      p.cF[b * kH + j] = cn;
      p.hF[b * kH + j] = hn;
      _Float16 hh = (_Float16)hn;
      p.Ag[b * kKih + 1024 + j] = hh;
      p.Av[b * kKih + j]        = hh;
    }
    grid_barrier(p.cnt, (++epoch) * kNWG);

    // ---- Phase B (WG0): attention logits, softmax, context ----
    if (wg == 0) {
      for (int e = tid; e < kB * kS; e += 256) {
        int b = e >> 6, s = e & 63;
        const float* sa = p.srcAtt + (size_t)(b * kS + s) * kH;
        const float* hv = p.hF + b * kH;
        float d = 0.f;
        for (int k = 0; k < kH; k++) d += sa[k] * hv[k];
        sl[e] = d;
      }
      __syncthreads();
      if (tid < kB) {
        float mx = -1e30f;
        for (int s = 0; s < kS; s++) mx = fmaxf(mx, sl[tid * kS + s]);
        float sum = 0.f;
        for (int s = 0; s < kS; s++) {
          float ev = __expf(sl[tid * kS + s] - mx);
          sl[tid * kS + s] = ev; sum += ev;
        }
        float inv = 1.0f / sum;
        for (int s = 0; s < kS; s++) sl[tid * kS + s] *= inv;
      }
      __syncthreads();
      for (int e = tid; e < kB * 1024; e += 256) {
        int b = e >> 10, k = e & 1023;
        const float* aw = sl + b * kS;
        const float* sv = p.srcT32 + (size_t)b * kS * 1024 + k;
        float a2 = 0.f;
        for (int s = 0; s < kS; s++) a2 += aw[s] * sv[(size_t)s * 1024];
        p.Av[b * kKih + 512 + k] = (_Float16)a2;
      }
      if (t + 1 < kT) {
        const _Float16* em = p.embF + (size_t)(t + 1) * kB * kE;
        for (int e = tid; e < kB * kE; e += 256) {
          int b = e >> 9, k = e & 511;
          p.Ag[b * kKih + k] = em[e];
        }
      }
    }
    grid_barrier(p.cnt, (++epoch) * kNWG);

    // ---- Phase C: att slice = tanh( Av(32x1536) * attW[n-slice]^T ) ----
    {
      const int mt = wave & 1, nt = (wave >> 1) & 3;
      v8f acc = {};
      const int n0 = j0 + nt * 16;
      for (int k0 = 0; k0 < kKih; k0 += 32) {
        v16h a = fragA_g(p.Av, lane, mt * 16, k0, kKih);
        v16h b = fragB_g(p.attW16, lane, n0, k0, kKih);
        acc = wmma16(a, b, acc);
      }
      int col = n0 + (lane & 15);
      int rowb = mt * 16 + ((lane >> 4) & 1) * 8;
#pragma unroll
      for (int r = 0; r < 8; r++) {
        int b = rowb + r;
        float v = tanhf(acc[r]);
        _Float16 hv = (_Float16)v;
        p.attAll[(size_t)(t * kB + b) * kH + col] = hv;
        p.Ag[b * kKih + 512 + col] = hv;
      }
    }
    grid_barrier(p.cnt, (++epoch) * kNWG);
  }
}

// ------------------------------ launch --------------------------------------

extern "C" void kernel_launch(void* const* d_in, const int* in_sizes, int n_in,
                              void* d_out, int out_size, void* d_ws, size_t ws_size,
                              hipStream_t stream) {
  Prms p;
  p.src     = (const float*)d_in[0];
  p.h0      = (const float*)d_in[1];
  p.c0      = (const float*)d_in[2];
  p.embTab  = (const float*)d_in[3];
  p.Wih     = (const float*)d_in[4];
  p.Whh     = (const float*)d_in[5];
  p.bih     = (const float*)d_in[6];
  p.bhh     = (const float*)d_in[7];
  p.attSrcW = (const float*)d_in[8];
  p.attVecW = (const float*)d_in[9];
  p.rdW     = (const float*)d_in[10];
  p.tgt     = (const int*)d_in[11];
  p.out     = (float*)d_out;

  char* w = (char*)d_ws;
  size_t o = 0;
  auto take = [&](size_t bytes) { char* r = w + o; o = (o + bytes + 255) & ~(size_t)255; return r; };
  p.rdW16     = (_Float16*)take((size_t)kV * kH * 2);
  p.W2        = (_Float16*)take((size_t)kG4 * kKih * 2);
  p.attW16    = (_Float16*)take((size_t)kH * 1536 * 2);
  p.attSrcW16 = (_Float16*)take((size_t)kH * 1024 * 2);
  p.srcT16    = (_Float16*)take((size_t)kB * kS * 1024 * 2);
  p.srcT32    = (float*)   take((size_t)kB * kS * 1024 * 4);
  p.srcAtt    = (float*)   take((size_t)kB * kS * kH * 4);
  p.embF      = (_Float16*)take((size_t)kT * kB * kE * 2);
  p.attAll    = (_Float16*)take((size_t)kT * kB * kH * 2);
  p.Ag        = (_Float16*)take((size_t)kB * kKih * 2);
  p.Av        = (_Float16*)take((size_t)kB * kKih * 2);
  p.hF        = (float*)   take((size_t)kB * kH * 4);
  p.cF        = (float*)   take((size_t)kB * kH * 4);
  p.bsum      = (float*)   take((size_t)kG4 * 4);
  p.cnt       = (unsigned*)take(256);

  prep_kernel<<<1024, 256, 0, stream>>>(p);

  // src_att = srcT(2048x1024) * attSrcW(512x1024)^T
  dim3 g1(kH / 128, (kB * kS) / 128);
  gemm_f16_tn<<<g1, 256, 0, stream>>>(p.srcT16, p.attSrcW16, p.srcAtt,
                                      kB * kS, kH, 1024);

  recurrent_kernel<<<kNWG, 256, 0, stream>>>(p);

  // scores = attAll(2048x512) * rdW(32000x512)^T  -> d_out (T,B,V)
  dim3 g2(kV / 128, (kT * kB) / 128);
  gemm_f16_tn<<<g2, 256, 0, stream>>>(p.attAll, p.rdW16, p.out,
                                      kT * kB, kV, 512);
}
